// MoELayer_66254165508232
// MI455X (gfx1250) — compile-verified, matrix operand order checked by reading, not run on hardware
//
#include <hip/hip_runtime.h>
#include <hip/hip_bf16.h>

// ---------------------------------------------------------------------------
// MoE (top-2 of 8 experts), SwiGLU, f32 in/out.  T=8192, D=1024, I=2048, E=8.
// Pipeline:
//   router -> offsets -> {gather x -> bf16 xg, transpose+convert weights}
//   -> grouped GEMMs staged by TDM (tensor_load_to_lds) + bf16 WMMA
// ---------------------------------------------------------------------------

typedef __attribute__((ext_vector_type(16))) __bf16 v16bf;
typedef __attribute__((ext_vector_type(8)))  float  v8f;
typedef unsigned int v4u __attribute__((ext_vector_type(4)));
typedef unsigned int v8u __attribute__((ext_vector_type(8)));

#define DIMD  1024
#define DIMI  2048
#define NEXP  8
#define BM    128    // token rows per block tile
#define BN    64     // output cols per block tile
#define BK    32     // K per WMMA step (bf16)
#define LDA   48     // LDS row stride in halfs (96B; via TDM pad: 16DW tile row + 8DW pad)
#define A_H   (BM * LDA)   // halfs per A LDS buffer
#define B_H   (BN * LDA)   // halfs per B LDS buffer

#define WMMA_BF16(a, b, c) \
  __builtin_amdgcn_wmma_f32_16x16x32_bf16(false, (a), false, (b), (short)0, (c), false, false)

__device__ __forceinline__ unsigned short f2bf(float f) {
  union { float f; unsigned u; } v; v.f = f;
  unsigned r = v.u + 0x7FFFu + ((v.u >> 16) & 1u);   // round-to-nearest-even
  return (unsigned short)(r >> 16);
}

// ---------------------------------------------------------------------------
// TDM: DMA a 2D tile (tile_k x tile_n halfs, k contiguous in memory with row
// length `dim0` halfs) into LDS at lptr, padding each 64B tile row to 96B
// (LDA = 48 halfs).  D# per CDNA5 ISA.
// NOTE: lptr is passed INTO the asm as an operand so the shared-memory object
// escapes — otherwise LLVM proves LDS is never written and deletes the
// subsequent fragment loads (observed in round 2).
// ---------------------------------------------------------------------------
__device__ __forceinline__ void tdm_load_2d(unsigned short* lptr, const void* gptr,
                                            unsigned dim0, unsigned tile_k,
                                            unsigned tile_n)
{
  const unsigned lds_off = (unsigned)(size_t)lptr;     // low 32b of generic = LDS byte offset
  unsigned long long ga = (unsigned long long)(size_t)gptr;
  v4u g0;
  g0[0] = 1u;                                          // count=1, user descriptor
  g0[1] = lds_off;                                     // lds_addr (bytes)
  g0[2] = (unsigned)ga;                                // global_addr[31:0]
  g0[3] = (unsigned)((ga >> 32) & 0x01FFFFFFu)         // global_addr[56:32]
        | (2u << 30);                                  // type = 2 ("image")
  v8u g1;
  g1[0] = (1u << 16)                                   // data_size = 2 bytes
        | (1u << 20)                                   // pad_enable
        | (3u << 22)                                   // pad_interval: 16 DWORDs (=64B row)
        | (7u << 25);                                  // pad_amount:   8 DWORDs (=32B)
  g1[1] = (dim0 & 0xFFFFu) << 16;                      // tensor_dim0[15:0]
  g1[2] = ((dim0 >> 16) & 0xFFFFu)                     // tensor_dim0[31:16]
        | (0xFFFFu << 16);                             // tensor_dim1[15:0] (huge, no OOB)
  g1[3] = 0xFu | (tile_k << 16);                       // tensor_dim1[31:16], tile_dim0
  g1[4] = tile_n;                                      // tile_dim1 (tile_dim2 = 0)
  g1[5] = dim0;                                        // tensor_dim0_stride[31:0]
  g1[6] = 0u;                                          // stride0[47:32], dim1_stride[15:0]
  g1[7] = 0u;                                          // dim1_stride[47:16] (2D: unused)
  v4u g2 = {0u, 0u, 0u, 0u};                           // dims 2/3: unused (tile_dim3=0)
  v4u g3 = {0u, 0u, 0u, 0u};                           // dim 4: unused (tile_dim4=0)
  asm volatile("tensor_load_to_lds %0, %1, %2, %3"
               :: "s"(g0), "s"(g1), "s"(g2), "s"(g3),
                  "v"(lptr)                            // escape LDS object (unused in template)
               : "memory");
}

// Gather a 16x32 bf16 fragment from LDS (row stride LDA).  ISA 7.12.2:
// lanes 0-15: elems 0-7 -> K 0-7,  8-15 -> K 16-23; lanes 16-31: +8.
__device__ __forceinline__ v16bf load_frag(const unsigned short* t, int base) {
  const int lane = threadIdx.x & 31;
  const unsigned short* p = t + (base + (lane & 15)) * LDA + ((lane >> 4) << 3);
  union { v16bf v; unsigned short u[16]; } f;
#pragma unroll
  for (int j = 0; j < 8; ++j) f.u[j] = p[j];
#pragma unroll
  for (int j = 0; j < 8; ++j) f.u[8 + j] = p[16 + j];
  return f.v;
}

// ---------------------------------------------------------------------------
// Router: one wave per token; top-2 renormalized softmax = sigmoid(logit gap).
// ---------------------------------------------------------------------------
__global__ __launch_bounds__(256) void router_kernel(
    const float* __restrict__ x, const float* __restrict__ rw,
    const float* __restrict__ rb, int* counts, int* tok_list, float* tok_wt,
    int T)
{
  const int t    = (blockIdx.x * blockDim.x + threadIdx.x) >> 5;
  const int lane = threadIdx.x & 31;
  if (t >= T) return;
  const float* xr = x + (size_t)t * DIMD;
  float acc[NEXP];
#pragma unroll
  for (int e = 0; e < NEXP; ++e) acc[e] = 0.f;
  for (int d = lane; d < DIMD; d += 32) {
    const float xv = xr[d];
    const float* w = rw + d * NEXP;
#pragma unroll
    for (int e = 0; e < NEXP; ++e) acc[e] += xv * w[e];
  }
#pragma unroll
  for (int off = 16; off > 0; off >>= 1) {
#pragma unroll
    for (int e = 0; e < NEXP; ++e) acc[e] += __shfl_xor(acc[e], off, 32);
  }
  if (lane == 0) {
#pragma unroll
    for (int e = 0; e < NEXP; ++e) acc[e] += rb[e];
    int e1 = 0;
#pragma unroll
    for (int e = 1; e < NEXP; ++e) if (acc[e] > acc[e1]) e1 = e;   // ties -> lowest
    int e2 = (e1 == 0) ? 1 : 0;
#pragma unroll
    for (int e = 0; e < NEXP; ++e) if (e != e1 && acc[e] > acc[e2]) e2 = e;
    const float w1 = 1.f / (1.f + __expf(acc[e2] - acc[e1]));
    const int p1 = atomicAdd(&counts[e1], 1);
    tok_list[e1 * T + p1] = t; tok_wt[e1 * T + p1] = w1;
    const int p2 = atomicAdd(&counts[e2], 1);
    tok_list[e2 * T + p2] = t; tok_wt[e2 * T + p2] = 1.f - w1;
  }
}

__global__ void offsets_kernel(const int* __restrict__ counts, int* offsets) {
  int s = 0;
#pragma unroll
  for (int e = 0; e < NEXP; ++e) { offsets[e] = s; s += counts[e]; }
  offsets[NEXP] = s;
}

// ---------------------------------------------------------------------------
// Gather routed token rows into compact bf16 activation buffer xg[2T, D].
// One block per slot (grid = 2T exactly: every token has exactly 2 experts).
// ---------------------------------------------------------------------------
__global__ __launch_bounds__(256) void gather_x_kernel(
    const float* __restrict__ x, const int* __restrict__ offsets,
    const int* __restrict__ tok_list, unsigned short* __restrict__ xg, int T)
{
  const int slot = blockIdx.x;
  int e = 0;
#pragma unroll
  for (int i = 1; i < NEXP; ++i) if (slot >= offsets[i]) e = i;
  const int tok = tok_list[e * T + (slot - offsets[e])];
  const float* src = x + (size_t)tok * DIMD;
  unsigned short* dst = xg + (size_t)slot * DIMD;
  for (int d = threadIdx.x; d < DIMD; d += 256) dst[d] = f2bf(src[d]);
}

// ---------------------------------------------------------------------------
// Convert + transpose weights to bf16 [n][k] (k = reduction dim contiguous).
// z = tensor*8 + e;  tensor 0: gate [D,I]->[I,D], 1: up, 2: down [I,D]->[D,I].
// ---------------------------------------------------------------------------
__global__ __launch_bounds__(256) void wtrans_kernel(
    const float* __restrict__ gate_w, const float* __restrict__ up_w,
    const float* __restrict__ down_w,
    unsigned short* __restrict__ gwT, unsigned short* __restrict__ uwT,
    unsigned short* __restrict__ dwT)
{
  const int z = blockIdx.z, tsel = z >> 3, e = z & 7;
  const float* src; unsigned short* dst; int R, C;
  if (tsel == 0)      { src = gate_w + (size_t)e * DIMD * DIMI; dst = gwT + (size_t)e * DIMI * DIMD; R = DIMD; C = DIMI; }
  else if (tsel == 1) { src = up_w   + (size_t)e * DIMD * DIMI; dst = uwT + (size_t)e * DIMI * DIMD; R = DIMD; C = DIMI; }
  else                { src = down_w + (size_t)e * DIMI * DIMD; dst = dwT + (size_t)e * DIMD * DIMI; R = DIMI; C = DIMD; }
  const int r0 = blockIdx.x * 32, c0 = blockIdx.y * 32;
  if (r0 >= R || c0 >= C) return;
  __shared__ unsigned short t[32][33];
  const int c  = threadIdx.x & 31;
  const int rg = (threadIdx.x >> 5) * 4;
#pragma unroll
  for (int j = 0; j < 4; ++j)
    t[rg + j][c] = f2bf(src[(size_t)(r0 + rg + j) * C + c0 + c]);
  __syncthreads();
  const int rr = threadIdx.x & 31;
  const int cg = (threadIdx.x >> 5) * 4;
#pragma unroll
  for (int j = 0; j < 4; ++j)
    dst[(size_t)(c0 + cg + j) * R + r0 + rr] = t[rr][cg + j];
}

// ---------------------------------------------------------------------------
// MLP stage 1: h = silu(x@gate_w + b) * (x@up_w + b).  TDM-staged, 8 waves,
// BM=128 x BN=64 tile, double-buffered LDS, bf16 WMMA f32-accumulate.
// ---------------------------------------------------------------------------
__global__ __launch_bounds__(256) void mlp1_kernel(
    const unsigned short* __restrict__ xg,
    const unsigned short* __restrict__ gwT, const float* __restrict__ gate_b,
    const unsigned short* __restrict__ uwT, const float* __restrict__ up_b,
    const int* __restrict__ counts, const int* __restrict__ offsets,
    unsigned short* __restrict__ h)
{
  const int e     = blockIdx.z;
  const int cnt   = counts[e];
  const int tile0 = blockIdx.x * BM;
  if (tile0 >= cnt) return;                 // block-uniform: EXEC stays full
  const int n0    = blockIdx.y * BN;
  const int off_e = offsets[e];

  __shared__ unsigned short smem[2 * A_H + 4 * B_H];   // A x2, Bg x2, Bu x2 (48 KB)
  unsigned short* bufA  = smem;
  unsigned short* bufBg = smem + 2 * A_H;
  unsigned short* bufBu = smem + 2 * A_H + 2 * B_H;

  const int tid  = threadIdx.x;
  const int wave = tid >> 5;
  const int lane = tid & 31;
  const int m0   = wave * 16;

  const unsigned short* Ap  = xg  + (size_t)(off_e + tile0) * DIMD;
  const unsigned short* Bgp = gwT + ((size_t)e * DIMI + n0) * DIMD;
  const unsigned short* Bup = uwT + ((size_t)e * DIMI + n0) * DIMD;

  v8f accg[4], accu[4];
#pragma unroll
  for (int nt = 0; nt < 4; ++nt) { accg[nt] = 0.f; accu[nt] = 0.f; }

  const int NS = DIMD / BK;
  if (wave == 0) {                           // preload step 0 into buffer 0
    tdm_load_2d(bufA,  Ap,  DIMD, BK, BM);
    tdm_load_2d(bufBg, Bgp, DIMD, BK, BN);
    tdm_load_2d(bufBu, Bup, DIMD, BK, BN);
  }
  for (int s = 0; s < NS; ++s) {
    const int b = s & 1;
    if (wave == 0) {
      if (s + 1 < NS) {
        const int k1 = (s + 1) * BK;
        const int nb = (s + 1) & 1;
        tdm_load_2d(bufA  + nb * A_H, Ap  + k1, DIMD, BK, BM);
        tdm_load_2d(bufBg + nb * B_H, Bgp + k1, DIMD, BK, BN);
        tdm_load_2d(bufBu + nb * B_H, Bup + k1, DIMD, BK, BN);
        __builtin_amdgcn_s_wait_tensorcnt(3);   // step-s tiles complete (in-order)
      } else {
        __builtin_amdgcn_s_wait_tensorcnt(0);
      }
    }
    __syncthreads();                            // buffers for step s visible to all
    v16bf a = load_frag(bufA + b * A_H, m0);
#pragma unroll
    for (int nt = 0; nt < 4; ++nt) {
      v16bf bg = load_frag(bufBg + b * B_H, nt * 16);
      accg[nt] = WMMA_BF16(a, bg, accg[nt]);
      v16bf bu = load_frag(bufBu + b * B_H, nt * 16);
      accu[nt] = WMMA_BF16(a, bu, accu[nt]);
    }
    __syncthreads();                            // done reading before re-fill
  }

  // Epilogue: bias + SiLU(g)*u -> h.  C layout: VGPR i -> M=i / M=8+i.
  const int mhi = (lane >> 4) << 3;
#pragma unroll
  for (int nt = 0; nt < 4; ++nt) {
    const int col = n0 + nt * 16 + (lane & 15);
    const float gb = gate_b[e * DIMI + col];
    const float ub = up_b[e * DIMI + col];
#pragma unroll
    for (int i = 0; i < 8; ++i) {
      const int gr = tile0 + m0 + mhi + i;
      if (gr < cnt) {
        const float g = accg[nt][i] + gb;
        const float u = accu[nt][i] + ub;
        h[(size_t)(off_e + gr) * DIMI + col] = f2bf((g / (1.f + __expf(-g))) * u);
      }
    }
  }
}

// ---------------------------------------------------------------------------
// MLP stage 2: y[tok] += w * (h @ down_w + down_b).  Two commutative f32
// atomic adds per output element -> deterministic.
// ---------------------------------------------------------------------------
__global__ __launch_bounds__(256) void mlp2_kernel(
    const unsigned short* __restrict__ h,
    const unsigned short* __restrict__ dwT, const float* __restrict__ down_b,
    const int* __restrict__ counts, const int* __restrict__ offsets,
    const int* __restrict__ tok_list, const float* __restrict__ tok_wt,
    float* __restrict__ y, int T)
{
  const int e     = blockIdx.z;
  const int cnt   = counts[e];
  const int tile0 = blockIdx.x * BM;
  if (tile0 >= cnt) return;
  const int n0    = blockIdx.y * BN;
  const int off_e = offsets[e];

  __shared__ unsigned short smem[2 * A_H + 2 * B_H];   // 36 KB
  unsigned short* bufA = smem;
  unsigned short* bufB = smem + 2 * A_H;

  const int tid  = threadIdx.x;
  const int wave = tid >> 5;
  const int lane = tid & 31;
  const int m0   = wave * 16;

  const unsigned short* Ap = h   + (size_t)(off_e + tile0) * DIMI;
  const unsigned short* Bp = dwT + ((size_t)e * DIMD + n0) * DIMI;

  v8f acc[4];
#pragma unroll
  for (int nt = 0; nt < 4; ++nt) acc[nt] = 0.f;

  const int NS = DIMI / BK;
  if (wave == 0) {
    tdm_load_2d(bufA, Ap, DIMI, BK, BM);
    tdm_load_2d(bufB, Bp, DIMI, BK, BN);
  }
  for (int s = 0; s < NS; ++s) {
    const int b = s & 1;
    if (wave == 0) {
      if (s + 1 < NS) {
        const int k1 = (s + 1) * BK;
        const int nb = (s + 1) & 1;
        tdm_load_2d(bufA + nb * A_H, Ap + k1, DIMI, BK, BM);
        tdm_load_2d(bufB + nb * B_H, Bp + k1, DIMI, BK, BN);
        __builtin_amdgcn_s_wait_tensorcnt(2);
      } else {
        __builtin_amdgcn_s_wait_tensorcnt(0);
      }
    }
    __syncthreads();
    v16bf a = load_frag(bufA + b * A_H, m0);
#pragma unroll
    for (int nt = 0; nt < 4; ++nt) {
      v16bf bv = load_frag(bufB + b * B_H, nt * 16);
      acc[nt] = WMMA_BF16(a, bv, acc[nt]);
    }
    __syncthreads();
  }

  const int mhi = (lane >> 4) << 3;
#pragma unroll
  for (int nt = 0; nt < 4; ++nt) {
    const int col = n0 + nt * 16 + (lane & 15);
    const float db = down_b[e * DIMD + col];
#pragma unroll
    for (int i = 0; i < 8; ++i) {
      const int gr = tile0 + m0 + mhi + i;
      if (gr < cnt) {
        const int   tok = tok_list[e * T + gr];
        const float w   = tok_wt[e * T + gr];
        atomicAdd(&y[(size_t)tok * DIMD + col], w * (acc[nt][i] + db));
      }
    }
  }
}

// ---------------------------------------------------------------------------
extern "C" void kernel_launch(void* const* d_in, const int* in_sizes, int n_in,
                              void* d_out, int out_size, void* d_ws, size_t ws_size,
                              hipStream_t stream) {
  const float* x      = (const float*)d_in[0];
  const float* rw     = (const float*)d_in[1];
  const float* rb     = (const float*)d_in[2];
  const float* gate_w = (const float*)d_in[3];
  const float* gate_b = (const float*)d_in[4];
  const float* up_w   = (const float*)d_in[5];
  const float* up_b   = (const float*)d_in[6];
  const float* down_w = (const float*)d_in[7];
  const float* down_b = (const float*)d_in[8];
  float* y = (float*)d_out;

  const int T = in_sizes[0] / DIMD;       // 8192 tokens; 2T slots total

  // Workspace layout (256B-aligned sections)
  char* ws = (char*)d_ws;
  size_t off = 0;
  auto alloc = [&](size_t bytes) {
    size_t p = off; off = (off + bytes + 255) & ~(size_t)255; return p;
  };
  int*   counts   = (int*)(ws + alloc(NEXP * 4));
  int*   offsets  = (int*)(ws + alloc((NEXP + 1) * 4));
  int*   tok_list = (int*)(ws + alloc((size_t)NEXP * T * 4));
  float* tok_wt   = (float*)(ws + alloc((size_t)NEXP * T * 4));
  unsigned short* xg  = (unsigned short*)(ws + alloc((size_t)(2 * T + BM) * DIMD * 2));
  unsigned short* gwT = (unsigned short*)(ws + alloc((size_t)NEXP * DIMI * DIMD * 2));
  unsigned short* uwT = (unsigned short*)(ws + alloc((size_t)NEXP * DIMI * DIMD * 2));
  unsigned short* dwT = (unsigned short*)(ws + alloc((size_t)NEXP * DIMD * DIMI * 2));
  unsigned short* h   = (unsigned short*)(ws + alloc((size_t)(2 * T + BM) * DIMI * 2));

  hipMemsetAsync(counts, 0, NEXP * 4, stream);
  hipMemsetAsync(y, 0, (size_t)T * DIMD * sizeof(float), stream);

  router_kernel<<<T / 8, 256, 0, stream>>>(x, rw, rb, counts, tok_list, tok_wt, T);
  offsets_kernel<<<1, 1, 0, stream>>>(counts, offsets);
  gather_x_kernel<<<2 * T, 256, 0, stream>>>(x, offsets, tok_list, xg, T);

  dim3 gt(64, 64, 3 * NEXP);              // 32x32 transpose tiles, guarded
  wtrans_kernel<<<gt, 256, 0, stream>>>(gate_w, up_w, down_w, gwT, uwT, dwT);

  dim3 g1(T / BM, DIMI / BN, NEXP);       // worst-case tiles; early-exit on counts
  mlp1_kernel<<<g1, 256, 0, stream>>>(xg, gwT, gate_b, uwT, up_b,
                                      counts, offsets, h);
  dim3 g2(T / BM, DIMD / BN, NEXP);
  mlp2_kernel<<<g2, 256, 0, stream>>>(h, dwT, down_b, counts, offsets,
                                      tok_list, tok_wt, y, T);
}
